// istft_48309792145658
// MI455X (gfx1250) — matrix operand order repaired and checked by reading, not compiled, not run on hardware
//
#include <hip/hip_runtime.h>

// ISTFT: frames = (Rk@magn + Ik@phase + ac) / nfft ; overlap-add hop=512, nfft=1024.
// One bf16 WMMA GEMM (K = 512 real + 512 imag = 1024) fused with overlap-add via
// global f32 atomics. A operand (constant kernel matrix) pre-packed to the exact
// per-lane WMMA fragment layout in d_ws (2 MB, L2-resident). MI455X / gfx1250, wave32.

#define NFFT    1024
#define NFREQ   512
#define HOP     512
#define BATCH   16
#define TLEN    4096
#define OUT_LEN ((TLEN - 1) * HOP + NFFT)   // 2097664 per batch
#define KCHUNKS 32                          // K = 1024 in chunks of 32
#define APACK_UINT4S (64u * 32u * 32u * 2u) // [mBlock][kc][lane][2 x uint4] = 131072
#define APACK_BYTES  ((size_t)APACK_UINT4S * 16u)  // 2 MB

typedef __attribute__((ext_vector_type(16))) __bf16 v16bf;
typedef __attribute__((ext_vector_type(8)))  float  v8f;

__device__ __forceinline__ unsigned short f2bf(float x) {
    return __builtin_bit_cast(unsigned short, (__bf16)x);
}
// lo/hi pair -> one uint (compiler emits v_cvt_pk_bf16_f32)
__device__ __forceinline__ unsigned packbf(float lo, float hi) {
    return (unsigned)f2bf(lo) | ((unsigned)f2bf(hi) << 16);
}

__global__ void zero_out_kernel(float4* __restrict__ out, long long n4) {
    long long i = (long long)blockIdx.x * blockDim.x + threadIdx.x;
    if (i < n4) out[i] = make_float4(0.f, 0.f, 0.f, 0.f);
}

// Pack [Rk | Ik] (fp32) into bf16 A-fragments: index = ((mBlock*32 + kc)*32 + lane)*2 + half.
// A 16x32 bf16 layout: lanes 0-15 row M=lane15, K {0..7,16..23}; lanes 16-31 row M=lane15,
// K {8..15,24..31}. half 0 -> K base 0, half 1 -> K base 16.
__global__ void pack_a_kernel(const float* __restrict__ rk, const float* __restrict__ ik,
                              uint4* __restrict__ apack) {
    const int idx = blockIdx.x * 256 + threadIdx.x;       // 0 .. 131071
    const int half   = idx & 1;
    const int lane   = (idx >> 1) & 31;
    const int kc     = (idx >> 6) & 31;
    const int mBlock = idx >> 11;                         // 0..63
    const int lane15 = lane & 15;
    const int laneHi = lane >> 4;
    const int o      = mBlock * 16 + lane15;
    const int kBase  = kc * 32 + laneHi * 8 + half * 16;
    unsigned u[4];
#pragma unroll
    for (int j = 0; j < 4; ++j) {
        const int K = kBase + 2 * j;
        const float* src = (K < NFREQ) ? rk : ik;
        const int f = (K < NFREQ) ? K : (K - NFREQ);
        u[j] = packbf(src[(size_t)o * NFREQ + f], src[(size_t)o * NFREQ + f + 1]);
    }
    apack[idx] = make_uint4(u[0], u[1], u[2], u[3]);
}

template <bool PACKED_A>
__global__ __launch_bounds__(256)
void istft_wmma_kernel(const float* __restrict__ magn,
                       const float* __restrict__ phase,
                       const float* __restrict__ ac,
                       const float* __restrict__ rk,
                       const float* __restrict__ ik,
                       const uint4* __restrict__ apack,
                       float* __restrict__ out) {
    // X tile as [t][k-pair] uints: pitch 17 uints per t-row; double-buffered.
    __shared__ unsigned int sXT32[2][64 * 17];

    const int tid   = threadIdx.x;
    const int tBase = blockIdx.x * 64;     // t-tile
    const int b     = blockIdx.z;

    const int w      = tid >> 5;           // wave id 0..7 (wave32)
    const int lane   = tid & 31;
    const int laneHi = lane >> 4;
    const int lane15 = lane & 15;
    const int mBlock = blockIdx.y * 8 + w; // 16-row block of o

    v8f acc[4];
#pragma unroll
    for (int n = 0; n < 4; ++n)
#pragma unroll
        for (int r = 0; r < 8; ++r) acc[n][r] = 0.0f;

    // ---- X loader mapping: k-row pair kp (0..15), t-group g (0..15), 4 t-values each
    const int kp = tid >> 4;
    const int g  = tid & 15;
    const float* pM = magn  + ((size_t)b * NFREQ + 2 * kp) * TLEN + tBase + g * 4;
    const float* pP = phase + ((size_t)b * NFREQ + 2 * kp) * TLEN + tBase + g * 4;
    unsigned int* sxBuf[2];
    sxBuf[0] = &sXT32[0][(g * 4) * 17 + kp];
    sxBuf[1] = &sXT32[1][(g * 4) * 17 + kp];

    // ---- B-fragment LDS pointers (loop-invariant): pairs j at 8-bit ds offsets
    const int kb2 = laneHi ? 4 : 0;        // lanes 0-15: K {0..7,16..23}, lanes 16-31: K {8..15,24..31}
    const unsigned int* bptr[2][4];
#pragma unroll
    for (int c = 0; c < 2; ++c)
#pragma unroll
        for (int n = 0; n < 4; ++n)
            bptr[c][n] = &sXT32[c][(n * 16 + lane15) * 17 + kb2];

    // ---- A-fragment base (packed path): per-chunk stride = 32 B (imm offsets)
    const uint4* apBase = PACKED_A
        ? apack + ((size_t)mBlock * KCHUNKS * 32 + lane) * 2 : (const uint4*)nullptr;

    // ---- preload chunk 0 and stage to LDS buffer 0
    {
        const float4 q0 = ((const float4*)pM)[0];
        const float4 q1 = ((const float4*)(pM + TLEN))[0];
        sxBuf[0][0 * 17] = packbf(q0.x, q1.x);
        sxBuf[0][1 * 17] = packbf(q0.y, q1.y);
        sxBuf[0][2 * 17] = packbf(q0.z, q1.z);
        sxBuf[0][3 * 17] = packbf(q0.w, q1.w);
    }
    __syncthreads();

    for (int kc = 0; kc < KCHUNKS; ++kc) {
        const int cur = kc & 1;

        // ---- issue next chunk's X global loads early (software pipeline)
        float4 n0, n1;
        if (kc + 1 < KCHUNKS) {
            const int k1 = kc + 1;
            const float* p = (k1 < 16) ? (pM + (size_t)k1 * 32 * TLEN)
                                       : (pP + (size_t)(k1 - 16) * 32 * TLEN);
            n0 = ((const float4*)p)[0];
            n1 = ((const float4*)(p + TLEN))[0];
        }
        if (kc + 2 < KCHUNKS) {   // warm L2 two chunks ahead (global_prefetch_b8)
            const int k2 = kc + 2;
            const float* p = (k2 < 16) ? (pM + (size_t)k2 * 32 * TLEN)
                                       : (pP + (size_t)(k2 - 16) * 32 * TLEN);
            __builtin_prefetch(p, 0, 1);
        }

        // ---- A fragment
        union { unsigned int u[8]; v16bf v; uint4 q[2]; } af;
        if (PACKED_A) {
            af.q[0] = apBase[kc * 2];
            af.q[1] = apBase[kc * 2 + 1];
        } else {
            const int fAll = kc * 32;
            const float* ksrc = (fAll < NFREQ) ? rk : ik;
            const int f0 = (fAll < NFREQ) ? fAll : (fAll - NFREQ);
            const int o  = mBlock * 16 + lane15;
            const int kb = laneHi * 8;
#pragma unroll
            for (int j = 0; j < 4; ++j) {
                const float* p0 = ksrc + (size_t)o * NFREQ + f0 + kb + 2 * j;
                af.u[j]     = packbf(p0[0], p0[1]);
                af.u[4 + j] = packbf(p0[16], p0[17]);
            }
        }

        // ---- load ALL B fragments first (distinct regs -> no WMMA WAR nops)
        union Bf { unsigned int u[8]; v16bf v; } bfv[4];
#pragma unroll
        for (int n = 0; n < 4; ++n) {
            const unsigned int* p = bptr[cur][n];
#pragma unroll
            for (int j = 0; j < 8; ++j) bfv[n].u[j] = p[j + (j >= 4 ? 4 : 0)];
        }

        // ---- 4 x v_wmma_f32_16x16x32_bf16 back-to-back
#pragma unroll
        for (int n = 0; n < 4; ++n)
            acc[n] = __builtin_amdgcn_wmma_f32_16x16x32_bf16(
                false, af.v, false, bfv[n].v, (short)0, acc[n], false, false);

        // ---- stage next chunk into the other LDS buffer (fills WMMA co-exec slots)
        if (kc + 1 < KCHUNKS) {
            unsigned int* sx = sxBuf[cur ^ 1];
            sx[0 * 17] = packbf(n0.x, n1.x);
            sx[1 * 17] = packbf(n0.y, n1.y);
            sx[2 * 17] = packbf(n0.z, n1.z);
            sx[3 * 17] = packbf(n0.w, n1.w);
        }
        __syncthreads();
    }

    // ---- epilogue: frames = (acc + ac[b,t]) / nfft ; overlap-add out[b, t*hop + o] +=
    const float inv = 1.0f / (float)NFFT;
    float* outB = out + (size_t)b * OUT_LEN;
    const int oRow0 = mBlock * 16 + 8 * laneHi;   // C layout: VGPR r -> M = r (+8 lanes 16-31)
#pragma unroll
    for (int n = 0; n < 4; ++n) {
        const int t   = tBase + n * 16 + lane15;  // C: N = lane & 15
        const float a = ac[b * TLEN + t];
        float* op = outB + (size_t)t * HOP + oRow0;
#pragma unroll
        for (int r = 0; r < 8; ++r)
            atomicAdd(op + r, (acc[n][r] + a) * inv);
    }
}

extern "C" void kernel_launch(void* const* d_in, const int* in_sizes, int n_in,
                              void* d_out, int out_size, void* d_ws, size_t ws_size,
                              hipStream_t stream) {
    const float* magn  = (const float*)d_in[0];   // [16,1,512,4096]
    const float* phase = (const float*)d_in[1];   // [16,1,512,4096]
    const float* ac    = (const float*)d_in[2];   // [16,1,4096]
    const float* rk    = (const float*)d_in[3];   // [1024,512]
    const float* ik    = (const float*)d_in[4];   // [1024,512]
    float* out = (float*)d_out;                   // [16, 2097664]

    const long long n4 = (long long)BATCH * OUT_LEN / 4;
    zero_out_kernel<<<dim3((unsigned)((n4 + 255) / 256)), dim3(256), 0, stream>>>(
        (float4*)out, n4);

    dim3 grid(TLEN / 64, NFFT / 128, BATCH);      // (64, 8, 16)
    const bool useWs = (ws_size >= APACK_BYTES);
    if (useWs) {
        uint4* apack = (uint4*)d_ws;
        pack_a_kernel<<<dim3(APACK_UINT4S / 256), dim3(256), 0, stream>>>(rk, ik, apack);
        istft_wmma_kernel<true><<<grid, dim3(256), 0, stream>>>(
            magn, phase, ac, rk, ik, apack, out);
    } else {
        istft_wmma_kernel<false><<<grid, dim3(256), 0, stream>>>(
            magn, phase, ac, rk, ik, nullptr, out);
    }
}
